// PointNet2GlobalFuser_25005299598073
// MI455X (gfx1250) — compile-verified, matrix operand order checked by reading, not run on hardware
//
#include <hip/hip_runtime.h>
#include <stdint.h>

typedef __attribute__((ext_vector_type(16))) __bf16 v16bf;
typedef __attribute__((ext_vector_type(8)))  float  v8f;
typedef __attribute__((ext_vector_type(4)))  int    v4i;

#define EPSV 1e-5f
#define LDA 40   // LDS row stride in halfwords: 80B, 16B aligned, bank-conflict-free

#if defined(__gfx1250__) && __has_builtin(__builtin_amdgcn_global_load_async_to_lds_b128) && __has_builtin(__builtin_amdgcn_s_wait_asynccnt)
#define USE_ASYNC 1
#else
#define USE_ASYNC 0
#endif

__device__ __forceinline__ unsigned short f32_to_bf16(float f){
  unsigned u = __builtin_bit_cast(unsigned, f);
  unsigned r = u + 0x7FFFu + ((u >> 16) & 1u);   // round-to-nearest-even
  return (unsigned short)(r >> 16);
}
__device__ __forceinline__ float bf16_to_f32(unsigned short h){
  unsigned u = ((unsigned)h) << 16;
  return __builtin_bit_cast(float, u);
}

#if USE_ASYNC
typedef __attribute__((address_space(1))) v4i gv4i_t;
typedef __attribute__((address_space(3))) v4i lv4i_t;
__device__ __forceinline__ void async_cp16(const void* g, void* l){
  __builtin_amdgcn_global_load_async_to_lds_b128((gv4i_t*)g, (lv4i_t*)l, 0, 0);
}
#endif

// ---------------- farthest point sampling: one block per batch ----------------
__global__ void fps_kernel(const float* __restrict__ pts, int cstride, int N,
                           int npoint, int* __restrict__ fidx){
  const int b = blockIdx.x;
  const int t = threadIdx.x;           // 256 threads
  __shared__ float dist[4096];
  __shared__ float rv[256];
  __shared__ int   ri[256];
  __shared__ int   cur;
  const float* base = pts + (size_t)b * N * cstride;
  for (int n = t; n < N; n += 256) dist[n] = 1e10f;
  if (t == 0) cur = 0;
  __syncthreads();
  for (int it = 0; it < npoint; ++it){
    const int far = cur;
    if (t == 0) fidx[b * npoint + it] = far;
    const float cx = base[(size_t)far * cstride + 0];
    const float cy = base[(size_t)far * cstride + 1];
    const float cz = base[(size_t)far * cstride + 2];
    float bv = -1.0f; int bi = 0;
    for (int n = t; n < N; n += 256){
      const float dx = base[(size_t)n*cstride+0] - cx;
      const float dy = base[(size_t)n*cstride+1] - cy;
      const float dz = base[(size_t)n*cstride+2] - cz;
      const float d  = dx*dx + dy*dy + dz*dz;
      const float nd = fminf(dist[n], d);
      dist[n] = nd;
      if (nd > bv){ bv = nd; bi = n; }
    }
    rv[t] = bv; ri[t] = bi;
    __syncthreads();
    for (int s2 = 128; s2 > 0; s2 >>= 1){
      if (t < s2){
        if (rv[t+s2] > rv[t] || (rv[t+s2] == rv[t] && ri[t+s2] < ri[t])){
          rv[t] = rv[t+s2]; ri[t] = ri[t+s2];
        }
      }
      __syncthreads();
    }
    if (t == 0) cur = ri[0];
    __syncthreads();
  }
}

// ---------------- ball query: thread per (b,s); first nsample in index order --
__global__ void ballq_kernel(const float* __restrict__ pts, int cstride, int N,
                             int S, int nsamp, float r2,
                             const int* __restrict__ fidx, int* __restrict__ idx,
                             float* __restrict__ nxyz, int total){
  const int qd = blockIdx.x * 128 + threadIdx.x;
  if (qd >= total) return;
  const int b = qd / S;
  const int f = fidx[qd];
  const float* base = pts + (size_t)b * N * cstride;
  const float cx = base[(size_t)f*cstride+0];
  const float cy = base[(size_t)f*cstride+1];
  const float cz = base[(size_t)f*cstride+2];
  nxyz[qd*3+0] = cx; nxyz[qd*3+1] = cy; nxyz[qd*3+2] = cz;
  int cnt = 0, first = f; bool haveFirst = false;
  int* op = idx + (size_t)qd * nsamp;
  for (int n = 0; n < N && cnt < nsamp; ++n){
    const float dx = base[(size_t)n*cstride+0]-cx;
    const float dy = base[(size_t)n*cstride+1]-cy;
    const float dz = base[(size_t)n*cstride+2]-cz;
    const float d  = dx*dx + dy*dy + dz*dz;
    if (d <= r2){
      if (!haveFirst){ first = n; haveFirst = true; }
      op[cnt++] = n;
    }
  }
  for (int k2 = cnt; k2 < nsamp; ++k2) op[k2] = first;
}

// ---------------- build grouped GEMM operand A (bf16, K padded) ---------------
__global__ void build_A(const float* __restrict__ coords, int cstride,
                        const float* __restrict__ nxyz,
                        const void* __restrict__ points, int ptsBf16,
                        const int* __restrict__ idx, int S, int nsamp, int Cpts,
                        int Kpad, int Nsrc, size_t total,
                        unsigned short* __restrict__ A){
  const size_t e = (size_t)blockIdx.x * 256 + threadIdx.x;
  if (e >= total) return;
  const int c = (int)(e % Kpad);
  const size_t m = e / Kpad;
  const int k = (int)(m % nsamp);
  const size_t bs = m / nsamp;
  const int b = (int)(bs / S);
  const int j = idx ? idx[bs * nsamp + k] : k;
  float val = 0.f;
  if (c < 3){
    val = coords[((size_t)b*Nsrc + j)*cstride + c];
    if (nxyz) val -= nxyz[bs*3 + c];
  } else if (c < 3 + Cpts){
    const size_t pe = ((size_t)b*Nsrc + j)*Cpts + (c - 3);
    val = ptsBf16 ? bf16_to_f32(((const unsigned short*)points)[pe])
                  : ((const float*)points)[pe];
  }
  A[e] = f32_to_bf16(val);
}

// ---------------- pack fp32 weights -> bf16 (512 x Kpad, zero pad) ------------
__global__ void pack_w(const float* __restrict__ W, int Cin, int Kpad,
                       unsigned short* __restrict__ out){
  const int e = blockIdx.x * 256 + threadIdx.x;
  if (e >= 512 * Kpad) return;
  const int c = e % Kpad, n = e / Kpad;
  out[e] = f32_to_bf16(c < Cin ? W[(size_t)n * Cin + c] : 0.f);
}

// ---------------- WMMA GEMM: Y[M,512] = A[M,K] * W[512,K]^T + bias ------------
// 256 threads = 8 waves; block tile 128(M) x 64(N); BK=32; wave tile 32x32.
// Async double-buffered LDS staging via GLOBAL_LOAD_ASYNC_TO_LDS (ASYNCcnt);
// synchronous staging fallback otherwise.
// Deterministic per-(Mblock,channel) partial sums for BatchNorm stats.
__global__ __launch_bounds__(256) void gemm_bf16(
    const unsigned short* __restrict__ A, int K,
    const unsigned short* __restrict__ W,
    const float* __restrict__ bias,
    unsigned short* __restrict__ Y,
    float* __restrict__ Psum, float* __restrict__ Psq){
  __shared__ unsigned short As[2][128 * LDA];
  __shared__ unsigned short Bs[2][64 * LDA];
  __shared__ float partS[4][64];
  __shared__ float partQ[4][64];

  const int t = threadIdx.x;
  const int lane = t & 31;
  const int wave = t >> 5;
  const int waveM = wave & 3;    // 4 waves tile M
  const int waveN = wave >> 2;   // 2 waves tile N
  const int rowSel = lane & 15;
  const int hi = lane >> 4;
  const int kb = hi * 8;         // K sub-offset per ISA 16-bit A/B layout
  const size_t mBase = (size_t)blockIdx.x * 128;
  const int nBase = blockIdx.y * 64;

  const int aRow0 = t >> 2,         aQ0 = (t & 3) * 8;
  const int aRow1 = (t + 256) >> 2, aQ1 = aQ0;        // (t+256)&3 == t&3
  const int bRow  = t >> 2,         bQ  = (t & 3) * 8;

  v8f zero;
  #pragma unroll
  for (int r = 0; r < 8; ++r) zero[r] = 0.f;
  v8f acc[2][2] = {{zero, zero}, {zero, zero}};

  union Frag { v16bf v; uint4 q[2]; };

#if USE_ASYNC
  // ---- CDNA5 async DMA double-buffered pipeline: one barrier per K-step ----
  {
    // prologue: tile 0 -> buffer 0
    async_cp16(&A[(mBase+aRow0)*K + aQ0],           &As[0][aRow0*LDA + aQ0]);
    async_cp16(&A[(mBase+aRow1)*K + aQ1],           &As[0][aRow1*LDA + aQ1]);
    async_cp16(&W[(size_t)(nBase+bRow)*K + bQ],     &Bs[0][bRow*LDA + bQ]);
  }
  int cur = 0;
  for (int k0 = 0; k0 < K; k0 += 32){
    __builtin_amdgcn_s_wait_asynccnt(0);  // this wave's tile-k0 DMA landed
    __syncthreads();                      // everyone's tile-k0 data published
    if (k0 + 32 < K){                     // overlap tile-(k0+32) DMA w/ compute
      const int nxt = cur ^ 1, kn = k0 + 32;
      async_cp16(&A[(mBase+aRow0)*K + kn + aQ0],       &As[nxt][aRow0*LDA + aQ0]);
      async_cp16(&A[(mBase+aRow1)*K + kn + aQ1],       &As[nxt][aRow1*LDA + aQ1]);
      async_cp16(&W[(size_t)(nBase+bRow)*K + kn + bQ], &Bs[nxt][bRow*LDA + bQ]);
    }
    const unsigned short* Ac = As[cur];
    const unsigned short* Bc = Bs[cur];
    Frag a[2], bb[2];
    #pragma unroll
    for (int i = 0; i < 2; ++i){
      const int r = waveM*32 + i*16 + rowSel;
      a[i].q[0] = *(const uint4*)&Ac[r*LDA + kb];
      a[i].q[1] = *(const uint4*)&Ac[r*LDA + kb + 16];
    }
    #pragma unroll
    for (int j = 0; j < 2; ++j){
      const int r = waveN*32 + j*16 + rowSel;
      bb[j].q[0] = *(const uint4*)&Bc[r*LDA + kb];
      bb[j].q[1] = *(const uint4*)&Bc[r*LDA + kb + 16];
    }
    #pragma unroll
    for (int i = 0; i < 2; ++i)
      #pragma unroll
      for (int j = 0; j < 2; ++j)
        acc[i][j] = __builtin_amdgcn_wmma_f32_16x16x32_bf16(
            false, a[i].v, false, bb[j].v, (short)0, acc[i][j], false, false);
    cur ^= 1;
  }
#else
  // ---- synchronous staging fallback ----
  for (int k0 = 0; k0 < K; k0 += 32){
    __syncthreads();
    *(uint4*)&As[0][aRow0*LDA + aQ0] = *(const uint4*)&A[(mBase+aRow0)*K + k0 + aQ0];
    *(uint4*)&As[0][aRow1*LDA + aQ1] = *(const uint4*)&A[(mBase+aRow1)*K + k0 + aQ1];
    *(uint4*)&Bs[0][bRow*LDA + bQ]   = *(const uint4*)&W[(size_t)(nBase+bRow)*K + k0 + bQ];
    if (k0 + 32 < K)
      __builtin_prefetch(&A[(mBase + aRow0)*K + k0 + 32], 0, 1);
    __syncthreads();

    Frag a[2], bb[2];
    #pragma unroll
    for (int i = 0; i < 2; ++i){
      const int r = waveM*32 + i*16 + rowSel;
      a[i].q[0] = *(const uint4*)&As[0][r*LDA + kb];
      a[i].q[1] = *(const uint4*)&As[0][r*LDA + kb + 16];
    }
    #pragma unroll
    for (int j = 0; j < 2; ++j){
      const int r = waveN*32 + j*16 + rowSel;
      bb[j].q[0] = *(const uint4*)&Bs[0][r*LDA + kb];
      bb[j].q[1] = *(const uint4*)&Bs[0][r*LDA + kb + 16];
    }
    #pragma unroll
    for (int i = 0; i < 2; ++i)
      #pragma unroll
      for (int j = 0; j < 2; ++j)
        acc[i][j] = __builtin_amdgcn_wmma_f32_16x16x32_bf16(
            false, a[i].v, false, bb[j].v, (short)0, acc[i][j], false, false);
  }
#endif

  // epilogue: bias add, bf16 store, deterministic per-channel sum/sumsq
  #pragma unroll
  for (int j = 0; j < 2; ++j){
    const int colL = waveN*32 + j*16 + rowSel;   // 0..63
    const int col  = nBase + colL;
    const float bv = bias[col];
    float s = 0.f, q = 0.f;
    #pragma unroll
    for (int i = 0; i < 2; ++i){
      const size_t mB = mBase + waveM*32 + i*16 + hi*8;
      #pragma unroll
      for (int r = 0; r < 8; ++r){
        const float y = acc[i][j][r] + bv;
        Y[(mB + r)*512 + col] = f32_to_bf16(y);
        s += y; q += y*y;
      }
    }
    s += __shfl_down(s, 16);
    q += __shfl_down(q, 16);
    if (hi == 0){ partS[waveM][colL] = s; partQ[waveM][colL] = q; }
  }
  __syncthreads();
  if (t < 64){
    const float s = partS[0][t] + partS[1][t] + partS[2][t] + partS[3][t];
    const float q = partQ[0][t] + partQ[1][t] + partQ[2][t] + partQ[3][t];
    Psum[(size_t)blockIdx.x*512 + nBase + t] = s;
    Psq [(size_t)blockIdx.x*512 + nBase + t] = q;
  }
}

// ---------------- BN stats finalize: scale/shift per channel ------------------
__global__ void bn_finalize(const float* __restrict__ Psum, const float* __restrict__ Psq,
                            int Mblocks, const float* __restrict__ g,
                            const float* __restrict__ beta, float invM,
                            float* __restrict__ scale, float* __restrict__ shift){
  const int c = threadIdx.x;  // 512
  float s = 0.f, q = 0.f;
  for (int mb = 0; mb < Mblocks; ++mb){
    s += Psum[(size_t)mb*512 + c];
    q += Psq [(size_t)mb*512 + c];
  }
  const float m  = s * invM;
  const float v  = q * invM - m*m;
  const float sc = g[c] * rsqrtf(v + EPSV);
  scale[c] = sc;
  shift[c] = beta[c] - m * sc;
}

// ---------------- elementwise BN+ReLU, 8 channels per thread ------------------
__global__ void bn_relu(const unsigned short* __restrict__ Y,
                        const float* __restrict__ scale, const float* __restrict__ shift,
                        size_t total8, unsigned short* __restrict__ X){
  const size_t e8 = (size_t)blockIdx.x * 256 + threadIdx.x;
  if (e8 >= total8) return;
  const size_t e = e8 * 8;
  const int c0 = (int)(e & 511);
  union V { uint4 q; unsigned short h[8]; } in, out;
  in.q = *(const uint4*)(Y + e);
  #pragma unroll
  for (int i = 0; i < 8; ++i){
    const float x = bf16_to_f32(in.h[i]) * scale[c0+i] + shift[c0+i];
    out.h[i] = f32_to_bf16(fmaxf(x, 0.f));
  }
  *(uint4*)(X + e) = out.q;
}

// ---------------- BN+ReLU fused with max-over-nsample pooling -----------------
__global__ void bn_relu_max(const unsigned short* __restrict__ Y,
                            const float* __restrict__ scale, const float* __restrict__ shift,
                            int nsamp, unsigned short* __restrict__ P){
  const size_t bs = blockIdx.x;
  const int t = threadIdx.x;  // 256, 2 channels each
  for (int cc = 0; cc < 2; ++cc){
    const int c = t + cc*256;
    const float sc = scale[c], sh = shift[c];
    float mx = 0.f;  // relu output >= 0
    for (int k = 0; k < nsamp; ++k){
      const float y = bf16_to_f32(Y[(bs*(size_t)nsamp + k)*512 + c]);
      mx = fmaxf(mx, fmaxf(y*sc + sh, 0.f));
    }
    P[bs*512 + c] = f32_to_bf16(mx);
  }
}

// ---------------- final FC head + batch-norm over B=4 -------------------------
__global__ void fc_head(const unsigned short* __restrict__ g,
                        const float* __restrict__ w, const float* __restrict__ bias,
                        const float* __restrict__ gam, const float* __restrict__ beta,
                        float* __restrict__ out){
  const int t = threadIdx.x;   // 128 = 4 batches x 32 outputs
  const int b = t >> 5, o = t & 31;
  float acc = bias[o];
  for (int c = 0; c < 512; ++c)
    acc += bf16_to_f32(g[b*512 + c]) * w[o*512 + c];
  const float h = fmaxf(acc, 0.f);
  __shared__ float hs[128];
  hs[t] = h;
  __syncthreads();
  float m = 0.f, q = 0.f;
  for (int bb = 0; bb < 4; ++bb){ const float v = hs[bb*32 + o]; m += v; q += v*v; }
  m *= 0.25f;
  const float v = q*0.25f - m*m;
  out[b*32 + o] = gam[o]*(h - m)*rsqrtf(v + EPSV) + beta[o];
}

extern "C" void kernel_launch(void* const* d_in, const int* in_sizes, int n_in,
                              void* d_out, int out_size, void* d_ws, size_t ws_size,
                              hipStream_t stream){
  (void)in_sizes; (void)n_in; (void)out_size; (void)ws_size;
  const float* xyz = (const float*)d_in[0];
  auto saw  = [&](int sa, int l){ return (const float*)d_in[1 + sa*12 + l]; };
  auto sab  = [&](int sa, int l){ return (const float*)d_in[1 + sa*12 + 3 + l]; };
  auto sag  = [&](int sa, int l){ return (const float*)d_in[1 + sa*12 + 6 + l]; };
  auto sabt = [&](int sa, int l){ return (const float*)d_in[1 + sa*12 + 9 + l]; };
  const float* fcw  = (const float*)d_in[37];
  const float* fcb  = (const float*)d_in[38];
  const float* fcg  = (const float*)d_in[39];
  const float* fcbt = (const float*)d_in[40];

  const int B = 4;
  const int N1 = 4096, S1 = 512, NS1 = 32;
  const int S2 = 128, NS2 = 64;
  const int NS3 = 128;
  const int M1 = B*S1*NS1, M2 = B*S2*NS2, M3 = B*NS3;

  char* base = (char*)d_ws; size_t off = 0;
  auto alloc = [&](size_t bytes)->void*{
    void* p = base + off;
    off += (bytes + 255) & ~(size_t)255;
    return p;
  };

  unsigned short* Wp[3][3];
  for (int sa = 0; sa < 3; ++sa)
    for (int l = 0; l < 3; ++l){
      const int Kp = (l == 0) ? 544 : 512;
      Wp[sa][l] = (unsigned short*)alloc((size_t)512*Kp*2);
    }
  float* Psum   = (float*)alloc((size_t)512*512*4);   // max Mblocks=512
  float* Psq    = (float*)alloc((size_t)512*512*4);
  float* scaleB = (float*)alloc(512*4);
  float* shiftB = (float*)alloc(512*4);
  int*   fidx1  = (int*)alloc((size_t)B*S1*4);
  int*   fidx2  = (int*)alloc((size_t)B*S2*4);
  float* nx1    = (float*)alloc((size_t)B*S1*3*4);
  float* nx2    = (float*)alloc((size_t)B*S2*3*4);
  int*   idx1   = (int*)alloc((size_t)B*S1*NS1*4);
  int*   idx2   = (int*)alloc((size_t)B*S2*NS2*4);
  unsigned short* bufA = (unsigned short*)alloc((size_t)M1*544*2);
  unsigned short* bufY = (unsigned short*)alloc((size_t)M1*512*2);
  unsigned short* P1   = (unsigned short*)alloc((size_t)B*S1*512*2);
  unsigned short* P2   = (unsigned short*)alloc((size_t)B*S2*512*2);
  unsigned short* P3   = (unsigned short*)alloc((size_t)B*512*2);

  for (int sa = 0; sa < 3; ++sa)
    for (int l = 0; l < 3; ++l){
      const int Cin = (l == 0) ? 515 : 512;
      const int Kp  = (l == 0) ? 544 : 512;
      pack_w<<<(512*Kp + 255)/256, 256, 0, stream>>>(saw(sa,l), Cin, Kp, Wp[sa][l]);
    }

  auto run_mlp = [&](int sa, int M, int nsamp, unsigned short* Pout){
    const int Mb = M / 128;
    for (int l = 0; l < 3; ++l){
      const int K = (l == 0) ? 544 : 512;
      gemm_bf16<<<dim3(Mb, 8), 256, 0, stream>>>(bufA, K, Wp[sa][l], sab(sa,l),
                                                 bufY, Psum, Psq);
      bn_finalize<<<1, 512, 0, stream>>>(Psum, Psq, Mb, sag(sa,l), sabt(sa,l),
                                         1.0f/(float)M, scaleB, shiftB);
      if (l < 2){
        const size_t tot8 = (size_t)M*64;   // 8 channels per thread
        bn_relu<<<(unsigned)((tot8 + 255)/256), 256, 0, stream>>>(bufY, scaleB, shiftB, tot8, bufA);
      } else {
        bn_relu_max<<<M/nsamp, 256, 0, stream>>>(bufY, scaleB, shiftB, nsamp, Pout);
      }
    }
  };

  // ---- SA1: coords = xyz[...,:3] (row stride 512), points = xyz (512 ch)
  fps_kernel<<<B, 256, 0, stream>>>(xyz, 512, N1, S1, fidx1);
  ballq_kernel<<<(B*S1 + 127)/128, 128, 0, stream>>>(xyz, 512, N1, S1, NS1,
                                                     0.2f*0.2f, fidx1, idx1, nx1, B*S1);
  { const size_t tot = (size_t)M1*544;
    build_A<<<(unsigned)((tot + 255)/256), 256, 0, stream>>>(
        xyz, 512, nx1, xyz, 0, idx1, S1, NS1, 512, 544, N1, tot, bufA); }
  run_mlp(0, M1, NS1, P1);

  // ---- SA2: coords = l1x (stride 3), points = l1p (bf16)
  fps_kernel<<<B, 256, 0, stream>>>(nx1, 3, S1, S2, fidx2);
  ballq_kernel<<<(B*S2 + 127)/128, 128, 0, stream>>>(nx1, 3, S1, S2, NS2,
                                                     0.4f*0.4f, fidx2, idx2, nx2, B*S2);
  { const size_t tot = (size_t)M2*544;
    build_A<<<(unsigned)((tot + 255)/256), 256, 0, stream>>>(
        nx1, 3, nx2, P1, 1, idx2, S2, NS2, 512, 544, S1, tot, bufA); }
  run_mlp(1, M2, NS2, P2);

  // ---- SA3: group_all (no subtraction, identity grouping)
  { const size_t tot = (size_t)M3*544;
    build_A<<<(unsigned)((tot + 255)/256), 256, 0, stream>>>(
        nx2, 3, nullptr, P2, 1, nullptr, 1, NS3, 512, 544, NS3, tot, bufA); }
  run_mlp(2, M3, NS3, P3);

  // ---- head
  fc_head<<<1, 128, 0, stream>>>(P3, fcw, fcb, fcg, fcbt, (float*)d_out);
}
